// LightGCN_learner_50379966382779
// MI455X (gfx1250) — matrix operand order repaired and checked by reading, not compiled, not run on hardware
//
#include <hip/hip_runtime.h>
#include <hip/hip_bf16.h>

#define N_USERS 6144
#define N_ITEMS 12288
#define N_NODES (N_USERS + N_ITEMS)
#define DIM 64
#define NNZ (N_NODES * 32)
#define KTOP 31   // K + 1

typedef __attribute__((ext_vector_type(16))) _Float16 v16h;
typedef __attribute__((ext_vector_type(8)))  _Float16 v8h;
typedef __attribute__((ext_vector_type(8)))  float    v8f;

union V16U { v16h v; v8h h[2]; };

// ---------------------------------------------------------------- fill zero
__global__ void lg_zero_f32(float* __restrict__ p, long n) {
    long i = (long)blockIdx.x * blockDim.x + threadIdx.x;
    long stride = (long)gridDim.x * blockDim.x;
    for (; i < n; i += stride) p[i] = 0.0f;
}

// ---------------------------------------------------------------- concat e0
__global__ void lg_concat(const float* __restrict__ u, const float* __restrict__ it,
                          float* __restrict__ e0) {
    long i = (long)blockIdx.x * blockDim.x + threadIdx.x;
    long n = (long)N_NODES * DIM;
    if (i >= n) return;
    long ub = (long)N_USERS * DIM;
    e0[i] = (i < ub) ? u[i] : it[i - ub];
}

// ---------------------------------------------------------------- SpMM (scatter, f32 atomics)
// thread t handles edge t>>4, dim chunk ((t&15)*4 .. +3)
__global__ void lg_spmm(const int* __restrict__ erow, const int* __restrict__ ecol,
                        const float* __restrict__ eval,
                        const float* __restrict__ x, float* __restrict__ y) {
    long tid = (long)blockIdx.x * blockDim.x + threadIdx.x;
    if (tid >= (long)NNZ * 16) return;
    int e = (int)(tid >> 4);
    int c = ((int)tid & 15) * 4;
    int r   = erow[e];
    int col = ecol[e];
    float v = eval[e];
    const float4 xv = *(const float4*)(x + (size_t)col * DIM + c);
    float* yp = y + (size_t)r * DIM + c;
    unsafeAtomicAdd(yp + 0, v * xv.x);
    unsafeAtomicAdd(yp + 1, v * xv.y);
    unsafeAtomicAdd(yp + 2, v * xv.z);
    unsafeAtomicAdd(yp + 3, v * xv.w);
}

// ---------------------------------------------------------------- combine + normalize -> f16
// one wave (32 lanes) per item row; each lane owns 2 dims
__global__ __launch_bounds__(256) void lg_combine(const float* __restrict__ e0,
                                                  const float* __restrict__ e1,
                                                  const float* __restrict__ e2,
                                                  _Float16* __restrict__ itemsN) {
    int wave = threadIdx.x >> 5;
    int lane = threadIdx.x & 31;
    int row  = blockIdx.x * 8 + wave;           // item index
    if (row >= N_ITEMS) return;
    size_t node = (size_t)(N_USERS + row) * DIM;
    int d0 = lane * 2;
    float a = (e0[node + d0]     + e1[node + d0]     + e2[node + d0])     * (1.0f / 3.0f);
    float b = (e0[node + d0 + 1] + e1[node + d0 + 1] + e2[node + d0 + 1]) * (1.0f / 3.0f);
    float ss = a * a + b * b;
    #pragma unroll
    for (int off = 16; off > 0; off >>= 1) ss += __shfl_xor(ss, off, 32);
    float nrm = sqrtf(ss);
    float scale = 1.0f / fmaxf(nrm, 1e-12f);
    size_t o = (size_t)row * DIM + d0;
    itemsN[o]     = (_Float16)(a * scale);
    itemsN[o + 1] = (_Float16)(b * scale);
}

// ---------------------------------------------------------------- WMMA Gram matrix
// one wave per 16x64 output strip: 1 M-tile x 4 N-tiles, K=64 (2 x wmma 16x16x32 f16)
__global__ __launch_bounds__(32) void lg_gram_wmma(const _Float16* __restrict__ A,
                                                   float* __restrict__ out) {
    const int lane  = threadIdx.x;
    const int half  = lane >> 4;     // 0 or 1
    const int l15   = lane & 15;
    const int tileM = blockIdx.y;                  // 0..767
    const int nBase = blockIdx.x * 4;              // 4 N-tiles per wave

    // A operand (row m = tileM*16 + l15):
    //   halves[0..7]  = A[m][koff .. koff+7],  halves[8..15] = A[m][koff+16 .. koff+23],
    //   koff = half*8 (+ kstep 0/32 for the two WMMAs)
    const int m = tileM * 16 + l15;
    const _Float16* arow = A + (size_t)m * DIM + half * 8;
    V16U a0, a1;
    a0.h[0] = *(const v8h*)(arow + 0);
    a0.h[1] = *(const v8h*)(arow + 16);
    a1.h[0] = *(const v8h*)(arow + 32);
    a1.h[1] = *(const v8h*)(arow + 48);

    #pragma unroll
    for (int t = 0; t < 4; ++t) {
        // B operand: B = A^T, lane holds col n = l15, same K layout as A (koff = half*8, +16)
        const int n = (nBase + t) * 16 + l15;
        const _Float16* brow = A + (size_t)n * DIM + half * 8;
        V16U b0, b1;
        b0.h[0] = *(const v8h*)(brow + 0);
        b0.h[1] = *(const v8h*)(brow + 16);
        b1.h[0] = *(const v8h*)(brow + 32);
        b1.h[1] = *(const v8h*)(brow + 48);

        v8f c = {};
        c = __builtin_amdgcn_wmma_f32_16x16x32_f16(false, a0.v, false, b0.v,
                                                   (short)0, c, false, false);
        c = __builtin_amdgcn_wmma_f32_16x16x32_f16(false, a1.v, false, b1.v,
                                                   (short)0, c, false, false);

        // C layout: VGPR r -> row tileM*16 + half*8 + r, col (nBase+t)*16 + l15
        float* orow = out + (size_t)(tileM * 16 + half * 8) * N_ITEMS
                          + (size_t)(nBase + t) * 16 + l15;
        #pragma unroll
        for (int r = 0; r < 8; ++r)
            orow[(size_t)r * N_ITEMS] = c[r];
    }
}

// ---------------------------------------------------------------- per-row top-31 + relu + sparsify
__global__ __launch_bounds__(256) void lg_topk(float* __restrict__ sim) {
    __shared__ float srow[N_ITEMS];      // 48 KB (LDS per WGP is 320 KB)
    __shared__ float rval[256];
    __shared__ int   ridx[256];
    __shared__ float kval[KTOP];
    __shared__ int   kidx[KTOP];

    const int tid = threadIdx.x;
    const size_t row = blockIdx.x;
    float* rp = sim + row * (size_t)N_ITEMS;

    for (int j = tid; j < N_ITEMS; j += 256) srow[j] = rp[j];
    __syncthreads();

    for (int it = 0; it < KTOP; ++it) {
        float best = -3.0e38f;
        int   bidx = 0;
        for (int j = tid; j < N_ITEMS; j += 256) {
            float v = srow[j];
            if (v > best) { best = v; bidx = j; }   // ascending scan -> lowest index on ties
        }
        rval[tid] = best; ridx[tid] = bidx;
        __syncthreads();
        #pragma unroll
        for (int s = 128; s > 0; s >>= 1) {
            if (tid < s) {
                float ov = rval[tid + s]; int oi = ridx[tid + s];
                if (ov > rval[tid] || (ov == rval[tid] && oi < ridx[tid])) {
                    rval[tid] = ov; ridx[tid] = oi;
                }
            }
            __syncthreads();
        }
        if (tid == 0) {
            kval[it] = rval[0];
            kidx[it] = ridx[0];
            srow[ridx[0]] = -3.0e38f;   // mask out
        }
        __syncthreads();
    }

    for (int j = tid; j < N_ITEMS; j += 256) rp[j] = 0.0f;
    __syncthreads();
    if (tid < KTOP) rp[kidx[tid]] = fmaxf(kval[tid], 0.0f);
}

// ---------------------------------------------------------------- host launcher
extern "C" void kernel_launch(void* const* d_in, const int* in_sizes, int n_in,
                              void* d_out, int out_size, void* d_ws, size_t ws_size,
                              hipStream_t stream) {
    const float* user_emb = (const float*)d_in[0];
    const float* item_emb = (const float*)d_in[1];
    const int*   edge_row = (const int*)d_in[2];
    const int*   edge_col = (const int*)d_in[3];
    const float* edge_val = (const float*)d_in[4];
    float* out = (float*)d_out;

    float* e0 = (float*)d_ws;
    float* e1 = e0 + (size_t)N_NODES * DIM;
    float* e2 = e1 + (size_t)N_NODES * DIM;
    _Float16* itemsN = (_Float16*)(e2 + (size_t)N_NODES * DIM);

    const long nodeElems = (long)N_NODES * DIM;

    // e0 = concat(user, item)
    lg_concat<<<(int)((nodeElems + 255) / 256), 256, 0, stream>>>(user_emb, item_emb, e0);

    // layer 1: e1 = A @ e0
    lg_zero_f32<<<2048, 256, 0, stream>>>(e1, nodeElems);
    {
        long nthr = (long)NNZ * 16;
        lg_spmm<<<(int)((nthr + 255) / 256), 256, 0, stream>>>(edge_row, edge_col, edge_val, e0, e1);
    }
    // layer 2: e2 = A @ e1
    lg_zero_f32<<<2048, 256, 0, stream>>>(e2, nodeElems);
    {
        long nthr = (long)NNZ * 16;
        lg_spmm<<<(int)((nthr + 255) / 256), 256, 0, stream>>>(edge_row, edge_col, edge_val, e1, e2);
    }

    // mean of 3 layers, row-normalize items, cast f16
    lg_combine<<<N_ITEMS / 8, 256, 0, stream>>>(e0, e1, e2, itemsN);

    // sim = itemsN @ itemsN^T  (f16 WMMA, f32 accum) -> d_out
    dim3 ggrid(N_ITEMS / 16 / 4, N_ITEMS / 16, 1);
    lg_gram_wmma<<<ggrid, 32, 0, stream>>>(itemsN, out);

    // per-row top-31, relu, sparsify in place
    lg_topk<<<N_ITEMS, 256, 0, stream>>>(out);
}
